// CustomAdapter_42099269435411
// MI455X (gfx1250) — matrix operand order repaired and checked by reading, compiled, not run on hardware
//
#include <hip/hip_runtime.h>
#include <hip/hip_bf16.h>

// MI455X / gfx1250: wave32, WMMA 16x16x32 bf16 -> f32.
// One wave per row of x[8192, 4096]. Memory-bound design: ~256MB HBM traffic
// (x read + out write; re-reads hit L2's 192MB), so the 8.6 GFLOP of tiny-GEMM
// work runs far below the bf16 WMMA ceiling. VALU per element minimized by
// using native f32->bf16 converts instead of manual RNE bit-twiddling.

typedef __attribute__((ext_vector_type(16))) __bf16 v16bf;
typedef __attribute__((ext_vector_type(8)))  float  v8f;

union FragU { v16bf v; unsigned int u[8]; };

#define DDIM 4096
#define NW 4          // waves (rows) per block
#define LSTR 66       // LDS row stride (bf16 elems) -> bank-conflict-free pair loads

__global__ __launch_bounds__(32 * NW) void adapter_kernel(
    const float* __restrict__ x, const float* __restrict__ Am,
    const float* __restrict__ Bm, const float* __restrict__ bias,
    const float* __restrict__ gamma, const float* __restrict__ beta,
    float* __restrict__ out, long long nrows) {
  __shared__ __bf16 sBt[64 * LSTR];        // B^T  row-major [q][p]  (A-operand, GEMM1)
  __shared__ __bf16 sAc[64 * LSTR];        // A    col-major [b][a]  (B-operand, GEMM2)
  __shared__ __bf16 sH [NW][64 * LSTR];    // H^T  [a][p]            (B-operand, GEMM1)
  __shared__ __bf16 sT [NW][16 * LSTR];    // T    [m][a] per mt     (A-operand, GEMM2)

  const int tid  = threadIdx.x;
  const int lane = tid & 31;
  const int w    = tid >> 5;
  const int n    = lane & 15;   // N index (B/C/D) or M index (A-operand)
  const int hi   = lane >> 4;   // lane-half

  // ---- one-time per block: constants -> bf16 LDS (transposed) ----
  for (int i = tid; i < 64 * 64; i += blockDim.x) {
    int r = i >> 6, c = i & 63;
    sBt[c * LSTR + r] = (__bf16)Bm[i];   // B[p=r][q=c] -> Bt[q][p]
    sAc[c * LSTR + r] = (__bf16)Am[i];   // A[a=r][b=c] -> Acm[b][a]
  }
  __syncthreads();

  const long long row = (long long)blockIdx.x * NW + w;
  if (row >= nrows) return;            // wave-uniform
  const float* xrow = x + row * DDIM;
  float*       orow = out + row * DDIM;

  // ---- LayerNorm pass 1: stats ----
  const float4* x4 = (const float4*)xrow;
  float sum = 0.f, ss = 0.f;
#pragma unroll
  for (int i = 0; i < 32; ++i) {
    float4 v = x4[i * 32 + lane];
    sum += v.x + v.y + v.z + v.w;
    ss  += v.x * v.x + v.y * v.y + v.z * v.z + v.w * v.w;
  }
#pragma unroll
  for (int off = 16; off > 0; off >>= 1) {
    sum += __shfl_xor(sum, off, 32);
    ss  += __shfl_xor(ss,  off, 32);
  }
  const float mu  = sum * (1.f / 4096.f);
  const float var = ss * (1.f / 4096.f) - mu * mu;
  const float rs  = rsqrtf(var + 1e-5f);

  // ---- LayerNorm pass 2: h -> bf16 (native cvt), stored transposed H^T[a][p] ----
  // d = 128*i + 4*lane  =>  a = 4*(lane&15) is lane-constant, p = 2*i + hi.
  __bf16* Hs = sH[w];
  __bf16* Ts = sT[w];
  const float4* g4 = (const float4*)gamma;
  const float4* b4 = (const float4*)beta;
  {
    __bf16* hbase = Hs + 4 * n * LSTR + hi;   // column p base for this lane
#pragma unroll
    for (int i = 0; i < 32; ++i) {
      int i4 = i * 32 + lane;
      float4 v = x4[i4]; float4 gg = g4[i4]; float4 bb = b4[i4];
      hbase[0 * LSTR + 2 * i] = (__bf16)((v.x - mu) * rs * gg.x + bb.x);
      hbase[1 * LSTR + 2 * i] = (__bf16)((v.y - mu) * rs * gg.y + bb.y);
      hbase[2 * LSTR + 2 * i] = (__bf16)((v.z - mu) * rs * gg.z + bb.z);
      hbase[3 * LSTR + 2 * i] = (__bf16)((v.w - mu) * rs * gg.w + bb.w);
    }
  }
  // Same-wave LDS: DS ops complete in order; compiler inserts dscnt waits.

  // ---- preload H as B-operand fragments [nt][ks] (reused across all mt) ----
  FragU hB[8];
#pragma unroll
  for (int nt = 0; nt < 4; ++nt)
#pragma unroll
    for (int ks = 0; ks < 2; ++ks) {
      FragU f;
#pragma unroll
      for (int j = 0; j < 8; ++j) {
        int k0 = ks * 32 + hi * 16 + 2 * j;
        f.u[j] = *(const unsigned int*)(Hs + (nt * 16 + n) * LSTR + k0);
      }
      hB[nt * 2 + ks] = f;
    }

#pragma unroll
  for (int mt = 0; mt < 4; ++mt) {
    // Bt A-operand fragments for this q-tile (both k-steps)
    FragU a0, a1;
#pragma unroll
    for (int j = 0; j < 8; ++j) {
      int kp = (hi << 3) + j;
      a0.u[j] = *(const unsigned int*)(sBt + (mt * 16 + n) * LSTR +      2 * kp);
      a1.u[j] = *(const unsigned int*)(sBt + (mt * 16 + n) * LSTR + 32 + 2 * kp);
    }

    // GEMM1: T[mt, nt] = (B^T H)[16x16] tiles
    v8f t[4];
#pragma unroll
    for (int nt = 0; nt < 4; ++nt) {
      v8f acc = {0.f, 0.f, 0.f, 0.f, 0.f, 0.f, 0.f, 0.f};
      acc = __builtin_amdgcn_wmma_f32_16x16x32_bf16(false, a0.v, false, hB[nt * 2 + 0].v,
                                                    (short)0, acc, false, false);
      acc = __builtin_amdgcn_wmma_f32_16x16x32_bf16(false, a1.v, false, hB[nt * 2 + 1].v,
                                                    (short)0, acc, false, false);
      t[nt] = acc;
    }

    // C/D layout -> row-major bf16 T[m][a] in LDS (same-wave relayout)
#pragma unroll
    for (int nt = 0; nt < 4; ++nt)
#pragma unroll
      for (int r = 0; r < 8; ++r)
        Ts[(r + 8 * hi) * LSTR + nt * 16 + n] = (__bf16)t[nt][r];

    // T as A-operand fragments
    FragU tA0, tA1;
#pragma unroll
    for (int j = 0; j < 8; ++j) {
      int kp = (hi << 3) + j;
      tA0.u[j] = *(const unsigned int*)(Ts + n * LSTR +      2 * kp);
      tA1.u[j] = *(const unsigned int*)(Ts + n * LSTR + 32 + 2 * kp);
    }

    // GEMM2: Y[mt, nt2] = T @ A, then fused epilogue
#pragma unroll
    for (int nt2 = 0; nt2 < 4; ++nt2) {
      FragU b0, b1;
#pragma unroll
      for (int j = 0; j < 8; ++j) {
        int k0 = hi * 16 + 2 * j;
        b0.u[j] = *(const unsigned int*)(sAc + (nt2 * 16 + n) * LSTR +      k0);
        b1.u[j] = *(const unsigned int*)(sAc + (nt2 * 16 + n) * LSTR + 32 + k0);
      }
      v8f y = {0.f, 0.f, 0.f, 0.f, 0.f, 0.f, 0.f, 0.f};
      y = __builtin_amdgcn_wmma_f32_16x16x32_bf16(false, tA0.v, false, b0.v,
                                                  (short)0, y, false, false);
      y = __builtin_amdgcn_wmma_f32_16x16x32_bf16(false, tA1.v, false, b1.v,
                                                  (short)0, y, false, false);
#pragma unroll
      for (int r = 0; r < 8; ++r) {
        int mm = r + 8 * hi;
        int d  = (mt * 16 + mm) * 64 + nt2 * 16 + n;
        float yv  = y[r];
        float gel = 0.5f * yv * (1.f + erff(yv * 0.70710678118654752f)); // exact GELU
        orow[d] = xrow[d] + gel + bias[d];                               // residual + bias
      }
    }
  }
}

extern "C" void kernel_launch(void* const* d_in, const int* in_sizes, int n_in,
                              void* d_out, int out_size, void* d_ws, size_t ws_size,
                              hipStream_t stream) {
  const float* x     = (const float*)d_in[0];
  const float* A     = (const float*)d_in[1];
  const float* B     = (const float*)d_in[2];
  const float* bias  = (const float*)d_in[3];
  const float* gamma = (const float*)d_in[4];
  const float* beta  = (const float*)d_in[5];
  float* out = (float*)d_out;

  long long nrows = (long long)in_sizes[0] / DDIM;   // 8192
  int blocks = (int)((nrows + NW - 1) / NW);         // 2048 blocks of 128 threads (4 waves)
  adapter_kernel<<<blocks, 32 * NW, 0, stream>>>(x, A, B, bias, gamma, beta, out, nrows);
}
